// LSTM_4183298146512
// MI455X (gfx1250) — matrix operand order repaired and checked by reading, not compiled, not run on hardware
//
#include <hip/hip_runtime.h>

// ---------------------------------------------------------------------------
// Types for CDNA5 WMMA (wave32): v_wmma_f32_16x16x32_f16
// ---------------------------------------------------------------------------
typedef __attribute__((ext_vector_type(16))) _Float16 v16h;
typedef __attribute__((ext_vector_type(8)))  _Float16 v8h;
typedef __attribute__((ext_vector_type(8)))  float    v8f;

#define Bn   256
#define Tn   336
#define Dn   64
#define Hn   256
#define G4n  1024   /* 4*H */

// CDNA5 async global->LDS staging (ASYNCcnt path), with safe fallback.
#if defined(__HIP_DEVICE_COMPILE__) && \
    __has_builtin(__builtin_amdgcn_global_load_async_to_lds_b128)
#define HAVE_ASYNC 1
#else
#define HAVE_ASYNC 0
#endif

#if HAVE_ASYNC
#define AS1 __attribute__((address_space(1)))
#define AS3 __attribute__((address_space(3)))
typedef int v4i_t __attribute__((vector_size(16)));   // b128 payload type
#endif

template <int N>
__device__ __forceinline__ void wait_async() {
#if HAVE_ASYNC
#if __has_builtin(__builtin_amdgcn_s_wait_asynccnt)
    __builtin_amdgcn_s_wait_asynccnt(N);
#else
    asm volatile("s_wait_asynccnt %0" :: "n"(N) : "memory");
#endif
#endif
}

// ---------------------------------------------------------------------------
// Per-direction scan parameters
// ---------------------------------------------------------------------------
struct ScanDir {
    const _Float16* in;     // [B][T][DIN] f16
    const _Float16* wih;    // [4H][DIN]   f16
    const _Float16* whh;    // [4H][H]     f16
    const float*    bias;   // [4H] combined b_ih + b_hh
    _Float16*       out16;  // [B][T][outRow] f16 hidden-seq sink (or nullptr)
    float*          hfin;   // [B][H] f32 final-h sink (or nullptr)
    int t0;                 // first timestep
    int dt;                 // +1 fwd / -1 rev
    int nsteps;             // number of steps to run
    int colOff;             // column offset into out16 rows
};

__device__ __forceinline__ v8f wmma16(v16h a, v16h b, v8f c) {
    // D = A(16x32,f16) x B(32x16,f16) + C(16x16,f32)
    return __builtin_amdgcn_wmma_f32_16x16x32_f16(
        /*neg_a=*/false, a, /*neg_b=*/false, b,
        /*c_mod=*/(short)0, c, /*reuse_a=*/false, /*reuse_b=*/false);
}

// A fragment from a row-major LDS tile [16][ldk] (f16).
// A layout matching B: lane<16 holds K = 32*kt .. +15, lanes>=16 hold
// K = 32*kt+16 .. +31.
__device__ __forceinline__ v16h load_a_lds(const _Float16* base, int ldk,
                                           int kt, int m, int hi) {
    const _Float16* p = base + m * ldk + kt * 32 + hi * 16;
    v8h lo = *(const v8h*)(p);
    v8h hh = *(const v8h*)(p + 8);
    return __builtin_shufflevector(lo, hh, 0, 1, 2, 3, 4, 5, 6, 7,
                                          8, 9, 10, 11, 12, 13, 14, 15);
}

// B fragment: column n of B == row (gate*H + n0 + col) of W ([4H][ldk] f16).
// Lane<16 holds K = 32*kt .. +15, lanes>=16 hold K = 32*kt+16 .. +31.
__device__ __forceinline__ v16h load_b_w(const _Float16* w, int ldk, int gate,
                                         int n0, int col, int hi, int kt) {
    const int grow = gate * Hn + n0 + col;
    return *(const v16h*)(w + (size_t)grow * ldk + kt * 32 + hi * 16);
}

__device__ __forceinline__ float sigf(float x) {
    return 1.0f / (1.0f + __expf(-x));
}

// ---------------------------------------------------------------------------
// Persistent LSTM scan: grid (B/16, 2 directions), 512 threads = 16 waves.
// Wave w owns h columns [16w, 16w+16); c stays in its accumulator registers
// for the whole scan; h round-trips through LDS. Input projection is fused
// (per-step K covers DIN then H). Input tiles are double-buffered and staged
// with async global->LDS loads overlapping the previous step's WMMA K-loop.
// ---------------------------------------------------------------------------
template <int DIN>
__global__ __launch_bounds__(512)
void lstm_scan_kernel(ScanDir d0, ScanDir d1, int T, int outRow) {
    __shared__ __align__(16) _Float16 s_in[2 * 16 * DIN]; // double-buffered
    __shared__ __align__(16) _Float16 s_h [16 * Hn];

    constexpr int CPR    = DIN / 8;            // 16B chunks per row
    constexpr int LOGCPR = (DIN == 64) ? 3 : 6;
    constexpr int CHUNKS = 16 * CPR;           // 16B chunks per tile
    constexpr int CPT    = (CHUNKS + 511) / 512;  // async ops per thread

    const ScanDir d = (blockIdx.y == 0) ? d0 : d1;
    const int tid  = threadIdx.x;
    const int wave = tid >> 5;
    const int lane = tid & 31;
    const int col  = lane & 15;        // N within 16x16 tile
    const int hi   = lane >> 4;        // lane half
    const int n0   = wave * 16;        // owned h-column base
    const int bt   = blockIdx.x * 16;  // batch tile base

    // Stage one input tile (16 rows x DIN f16) into buffer `buf` for time t.
    auto stage = [&](int buf, int t) {
        for (int c = tid; c < CHUNKS; c += 512) {
            const int row = c >> LOGCPR;
            const int j   = (c & (CPR - 1)) * 8;     // f16 offset in row
            const _Float16* g = d.in + ((size_t)(bt + row) * T + t) * DIN + j;
            _Float16* l = s_in + buf * (16 * DIN) + row * DIN + j;
#if HAVE_ASYNC
            __builtin_amdgcn_global_load_async_to_lds_b128(
                (AS1 v4i_t*)g, (AS3 v4i_t*)l, 0, 0);
#else
            *(uint4*)l = *(const uint4*)g;
#endif
        }
    };

    for (int i = tid; i < 16 * Hn; i += 512) s_h[i] = (_Float16)0.0f;

    // Column-broadcast gate biases (i,f,g,o), hoisted out of the scan.
    const float b0 = d.bias[0 * Hn + n0 + col];
    const float b1 = d.bias[1 * Hn + n0 + col];
    const float b2 = d.bias[2 * Hn + n0 + col];
    const float b3 = d.bias[3 * Hn + n0 + col];

    v8f cacc = {0.f, 0.f, 0.f, 0.f, 0.f, 0.f, 0.f, 0.f};  // cell state

    stage(0, d.t0);                 // prologue: stage first tile
    int buf = 0;

    for (int s = 0; s < d.nsteps; ++s) {
        const int t = d.t0 + s * d.dt;

        // Issue next tile's stage, then drain the current tile's async ops
        // (in-order completion: allowing CPT outstanding == previous done).
        if (s + 1 < d.nsteps) {
            stage(buf ^ 1, t + d.dt);
            wait_async<CPT>();
        } else {
            wait_async<0>();
        }
        __syncthreads();   // tile + previous h-writes visible to all waves

        const _Float16* cur = s_in + buf * (16 * DIN);

        v8f a0 = {b0, b0, b0, b0, b0, b0, b0, b0};
        v8f a1 = {b1, b1, b1, b1, b1, b1, b1, b1};
        v8f a2 = {b2, b2, b2, b2, b2, b2, b2, b2};
        v8f a3 = {b3, b3, b3, b3, b3, b3, b3, b3};

        // Fused input projection: g += x_t @ W_ih^T
        #pragma unroll 4
        for (int kt = 0; kt < (DIN >> 5); ++kt) {
            v16h af = load_a_lds(cur, DIN, kt, col, hi);
            a0 = wmma16(af, load_b_w(d.wih, DIN, 0, n0, col, hi, kt), a0);
            a1 = wmma16(af, load_b_w(d.wih, DIN, 1, n0, col, hi, kt), a1);
            a2 = wmma16(af, load_b_w(d.wih, DIN, 2, n0, col, hi, kt), a2);
            a3 = wmma16(af, load_b_w(d.wih, DIN, 3, n0, col, hi, kt), a3);
        }
        // Recurrent part: g += h_{t-1} @ W_hh^T
        #pragma unroll
        for (int kt = 0; kt < (Hn >> 5); ++kt) {
            v16h af = load_a_lds(s_h, Hn, kt, col, hi);
            a0 = wmma16(af, load_b_w(d.whh, Hn, 0, n0, col, hi, kt), a0);
            a1 = wmma16(af, load_b_w(d.whh, Hn, 1, n0, col, hi, kt), a1);
            a2 = wmma16(af, load_b_w(d.whh, Hn, 2, n0, col, hi, kt), a2);
            a3 = wmma16(af, load_b_w(d.whh, Hn, 3, n0, col, hi, kt), a3);
        }

        // Gate nonlinearity + state update (pure lane-local f32).
        v8f hh;
        #pragma unroll
        for (int r = 0; r < 8; ++r) {
            float iv = sigf(a0[r]);
            float fv = sigf(a1[r]);
            float gv = tanhf(a2[r]);
            float ov = sigf(a3[r]);
            float cn = fv * cacc[r] + iv * gv;
            cacc[r] = cn;
            hh[r] = ov * tanhf(cn);
        }

        __syncthreads();   // all waves done reading s_h / s_in[buf]

        #pragma unroll
        for (int r = 0; r < 8; ++r) {
            const int row = r + hi * 8;           // C/D layout: M = r + 8*hi
            const _Float16 h16 = (_Float16)hh[r];
            s_h[row * Hn + n0 + col] = h16;
            if (d.out16)
                d.out16[((size_t)(bt + row) * T + t) * outRow +
                        d.colOff + n0 + col] = h16;
        }
        if (d.hfin && s == d.nsteps - 1) {
            #pragma unroll
            for (int r = 0; r < 8; ++r) {
                const int row = r + hi * 8;
                d.hfin[(size_t)(bt + row) * Hn + n0 + col] = hh[r];
            }
        }
        buf ^= 1;
        // next iteration's pre-K barrier orders these s_h writes vs. reads
    }
}

// ---------------------------------------------------------------------------
// Helpers: f32 -> f16 convert, bias combine, FC head
// ---------------------------------------------------------------------------
__global__ void cvt_f16_kernel(const float* __restrict__ src,
                               _Float16* __restrict__ dst, int n) {
    int i = blockIdx.x * blockDim.x + threadIdx.x;
    if (i < n) dst[i] = (_Float16)src[i];
}

__global__ void bias_comb_kernel(const float* __restrict__ a,
                                 const float* __restrict__ b,
                                 float* __restrict__ o, int n) {
    int i = blockIdx.x * blockDim.x + threadIdx.x;
    if (i < n) o[i] = a[i] + b[i];
}

__global__ void fc_kernel(const float* __restrict__ hf,
                          const float* __restrict__ hr,
                          const float* __restrict__ w,   // [1][512]
                          const float* __restrict__ bb,  // [1]
                          float* __restrict__ out) {
    int b = blockIdx.x * blockDim.x + threadIdx.x;
    if (b < Bn) {
        float s = bb[0];
        for (int j = 0; j < Hn; ++j) s += hf[b * Hn + j] * w[j];
        for (int j = 0; j < Hn; ++j) s += hr[b * Hn + j] * w[Hn + j];
        out[b] = s;
    }
}

// ---------------------------------------------------------------------------
// Launch
// ---------------------------------------------------------------------------
extern "C" void kernel_launch(void* const* d_in, const int* in_sizes, int n_in,
                              void* d_out, int out_size, void* d_ws, size_t ws_size,
                              hipStream_t stream) {
    (void)in_sizes; (void)n_in; (void)out_size; (void)ws_size;

    const float* x       = (const float*)d_in[0];
    const float* wih_l0  = (const float*)d_in[1];
    const float* whh_l0  = (const float*)d_in[2];
    const float* bih_l0  = (const float*)d_in[3];
    const float* bhh_l0  = (const float*)d_in[4];
    const float* wih_l0r = (const float*)d_in[5];
    const float* whh_l0r = (const float*)d_in[6];
    const float* bih_l0r = (const float*)d_in[7];
    const float* bhh_l0r = (const float*)d_in[8];
    const float* wih_l1  = (const float*)d_in[9];
    const float* whh_l1  = (const float*)d_in[10];
    const float* bih_l1  = (const float*)d_in[11];
    const float* bhh_l1  = (const float*)d_in[12];
    const float* wih_l1r = (const float*)d_in[13];
    const float* whh_l1r = (const float*)d_in[14];
    const float* bih_l1r = (const float*)d_in[15];
    const float* bhh_l1r = (const float*)d_in[16];
    const float* fc_w    = (const float*)d_in[17];
    const float* fc_b    = (const float*)d_in[18];

    // --- workspace carve-out (256B aligned) ---
    char* ws = (char*)d_ws;
    size_t off = 0;
    auto carve = [&](size_t bytes) -> void* {
        off = (off + 255) & ~(size_t)255;
        void* p = ws + off;
        off += bytes;
        return p;
    };
    _Float16* x16   = (_Float16*)carve((size_t)Bn * Tn * Dn * 2);
    _Float16* wih0f = (_Float16*)carve((size_t)G4n * Dn * 2);
    _Float16* whh0f = (_Float16*)carve((size_t)G4n * Hn * 2);
    _Float16* wih0r = (_Float16*)carve((size_t)G4n * Dn * 2);
    _Float16* whh0r = (_Float16*)carve((size_t)G4n * Hn * 2);
    _Float16* wih1f = (_Float16*)carve((size_t)G4n * 2 * Hn * 2);
    _Float16* whh1f = (_Float16*)carve((size_t)G4n * Hn * 2);
    _Float16* wih1r = (_Float16*)carve((size_t)G4n * 2 * Hn * 2);
    _Float16* whh1r = (_Float16*)carve((size_t)G4n * Hn * 2);
    float*    bias0f = (float*)carve((size_t)G4n * 4);
    float*    bias0r = (float*)carve((size_t)G4n * 4);
    float*    bias1f = (float*)carve((size_t)G4n * 4);
    float*    bias1r = (float*)carve((size_t)G4n * 4);
    _Float16* hseq0  = (_Float16*)carve((size_t)Bn * Tn * 2 * Hn * 2); // [B][T][512]
    float*    h1f    = (float*)carve((size_t)Bn * Hn * 4);
    float*    h1r    = (float*)carve((size_t)Bn * Hn * 4);

    // --- f32 -> f16 conversions ---
    struct CvtJob { const float* s; _Float16* d; int n; };
    const CvtJob jobs[] = {
        { x,       x16,   Bn * Tn * Dn },
        { wih_l0,  wih0f, G4n * Dn     },
        { whh_l0,  whh0f, G4n * Hn     },
        { wih_l0r, wih0r, G4n * Dn     },
        { whh_l0r, whh0r, G4n * Hn     },
        { wih_l1,  wih1f, G4n * 2 * Hn },
        { whh_l1,  whh1f, G4n * Hn     },
        { wih_l1r, wih1r, G4n * 2 * Hn },
        { whh_l1r, whh1r, G4n * Hn     },
    };
    for (const auto& j : jobs)
        cvt_f16_kernel<<<(j.n + 255) / 256, 256, 0, stream>>>(j.s, j.d, j.n);

    bias_comb_kernel<<<G4n / 256, 256, 0, stream>>>(bih_l0,  bhh_l0,  bias0f, G4n);
    bias_comb_kernel<<<G4n / 256, 256, 0, stream>>>(bih_l0r, bhh_l0r, bias0r, G4n);
    bias_comb_kernel<<<G4n / 256, 256, 0, stream>>>(bih_l1,  bhh_l1,  bias1f, G4n);
    bias_comb_kernel<<<G4n / 256, 256, 0, stream>>>(bih_l1r, bhh_l1r, bias1r, G4n);

    // --- Layer 0: fwd + rev concurrently; write f16 hidden seq [B][T][512] ---
    ScanDir l0f = { x16, wih0f, whh0f, bias0f, hseq0, nullptr, 0,      +1, Tn, 0  };
    ScanDir l0r = { x16, wih0r, whh0r, bias0r, hseq0, nullptr, Tn - 1, -1, Tn, Hn };
    lstm_scan_kernel<Dn><<<dim3(Bn / 16, 2), 512, 0, stream>>>(l0f, l0r, Tn, 2 * Hn);

    // --- Layer 1: only h[:, T-1, :] is consumed downstream.
    //     fwd: full scan, keep final h.  rev: exactly one step at t = T-1. ---
    ScanDir l1f = { hseq0, wih1f, whh1f, bias1f, nullptr, h1f, 0,      +1, Tn, 0 };
    ScanDir l1r = { hseq0, wih1r, whh1r, bias1r, nullptr, h1r, Tn - 1, -1, 1,  0 };
    lstm_scan_kernel<2 * Hn><<<dim3(Bn / 16, 2), 512, 0, stream>>>(l1f, l1r, Tn, 0);

    // --- FC head: out[b] = [h1f | h1r] . fc_w + fc_b ---
    fc_kernel<<<1, Bn, 0, stream>>>(h1f, h1r, fc_w, fc_b, (float*)d_out);
}
